// Net_dnc_71957882077586
// MI455X (gfx1250) — compile-verified, matrix operand order checked by reading, not run on hardware
//
#include <hip/hip_runtime.h>

#define SEQ      8192
#define EMBED    20
#define HID      20
#define GATES    80          // 4*HID
#define CTRL     64
#define MEM_H    16
#define MEM_W    16
#define N_READ   4
#define NUM_ACT  1000
#define NUM_SYM  1000000
#define EPSF     1e-6f
#define CLIPV    20.0f

typedef __attribute__((ext_vector_type(16))) _Float16 v16h;
typedef __attribute__((ext_vector_type(8)))  float    v8f;

__device__ __forceinline__ float fsig(float x)   { return __builtin_amdgcn_rcpf(1.0f + __expf(-x)); }
__device__ __forceinline__ float ftanh(float x)  { return 1.0f - 2.0f * __builtin_amdgcn_rcpf(__expf(2.0f * x) + 1.0f); }
__device__ __forceinline__ float fsoftp(float x) { return (x > 20.0f) ? x : __logf(1.0f + __expf(x)); }

// ---------------------------------------------------------------------------
// K1: pre-gates = (emb[x]*mask) @ Wih^T + bih + bhh   via v_wmma_f32_16x16x32_f16
// One wave per 16-row time tile; K=20 zero-padded to 32 -> one WMMA per N-tile.
// ---------------------------------------------------------------------------
__global__ void k_pregate(const int* __restrict__ x, const float* __restrict__ emb,
                          const float* __restrict__ Wih, const float* __restrict__ bih,
                          const float* __restrict__ bhh, float* __restrict__ pre)
{
    const int lane = threadIdx.x & 31;
    const int wave = threadIdx.x >> 5;
    const int tile = blockIdx.x * 8 + wave;        // 0..511
    const int t0   = tile * 16;
    const int row  = lane & 15;
    const int hi   = lane >> 4;                    // half-wave selector
    const int t    = t0 + row;

    const int    idx   = x[t];
    const float  maskf = (idx != NUM_SYM) ? 1.0f : 0.0f;
    const float* erow  = emb + (size_t)idx * EMBED;

    // A fragment: 16x32 f16, lane layout per CDNA5 ISA (k = (e<8?e:e+8) + hi*8)
    v16h a;
#pragma unroll
    for (int e = 0; e < 16; ++e) {
        int   k = (e < 8 ? e : e + 8) + hi * 8;
        float v = erow[k < EMBED ? k : 0];
        a[e] = (_Float16)((k < EMBED) ? v * maskf : 0.0f);
    }

    const int col = lane & 15;
#pragma unroll
    for (int nt = 0; nt < 5; ++nt) {               // 80 gate columns = 5 N-tiles
        const int n = nt * 16 + col;
        v16h b;                                    // B fragment: 32x16, B[k][n] = Wih[n][k]
#pragma unroll
        for (int e = 0; e < 16; ++e) {
            int   k = (e < 8 ? e : e + 8) + hi * 8;
            float v = Wih[n * EMBED + (k < EMBED ? k : 0)];
            b[e] = (_Float16)((k < EMBED) ? v : 0.0f);
        }
        v8f c = {};
        c = __builtin_amdgcn_wmma_f32_16x16x32_f16(false, a, false, b,
                                                   (short)0, c, false, false);
        const float bias = bih[n] + bhh[n];
#pragma unroll
        for (int d = 0; d < 8; ++d) {
            int m = d + hi * 8;                    // D layout: VGPR d -> M = d (+8 for hi half)
            pre[(size_t)(t0 + m) * GATES + n] = c[d] + bias;
        }
    }
}

// ---------------------------------------------------------------------------
// K2: sequential LSTM recurrence, one wave. Whh pinned in VGPRs (80/lane),
// h broadcast via LDS (single-wave workgroup: barriers are S_NOP, DS in-order).
// ---------------------------------------------------------------------------
__global__ void __launch_bounds__(32)
k_lstm(const float* __restrict__ pre, const float* __restrict__ Whh,
       float* __restrict__ x4)
{
    const int tid = threadIdx.x;                   // 0..31
    const int j   = tid < HID ? tid : HID - 1;     // lanes >=20 mirror lane 19 (no OOB)

    float wi[HID], wf[HID], wg[HID], wo[HID];
#pragma unroll
    for (int k = 0; k < HID; ++k) {
        wi[k] = Whh[(0 * HID + j) * HID + k];
        wf[k] = Whh[(1 * HID + j) * HID + k];
        wg[k] = Whh[(2 * HID + j) * HID + k];
        wo[k] = Whh[(3 * HID + j) * HID + k];
    }

    __shared__ float lh[HID];
    float hall[HID];
#pragma unroll
    for (int k = 0; k < HID; ++k) hall[k] = 0.0f;
    float c = 0.0f, s = 0.0f;

    for (int t = 0; t < SEQ; ++t) {
        const float* p = pre + (size_t)t * GATES;
        float gi = p[j], gf = p[HID + j], gg = p[2 * HID + j], go = p[3 * HID + j];

        // hide L2 latency on the pre-gate stream (global_prefetch_b8)
        int tp = (t + 16 < SEQ) ? (t + 16) : (SEQ - 1);
        __builtin_prefetch((const char*)(pre + (size_t)tp * GATES) + tid * 16, 0, 3);

#pragma unroll
        for (int k = 0; k < HID; ++k) {
            float h = hall[k];
            gi += h * wi[k]; gf += h * wf[k]; gg += h * wg[k]; go += h * wo[k];
        }
        float cn = fsig(gf) * c + fsig(gi) * ftanh(gg);
        float hn = fsig(go) * ftanh(cn);
        c = cn; s += hn;

        if (tid < HID) lh[tid] = hn;
        __syncthreads();
#pragma unroll
        for (int k = 0; k < HID; ++k) hall[k] = lh[k];
        __syncthreads();
    }
    if (tid < HID) x4[tid] = s;
}

// ---------------------------------------------------------------------------
// K3: single DNC step from reset state (read_prev=0, h0=c0=0, link=preced=0,
// usage stays 0 -> fw=bw=0, psi=1). One 256-thread block.
// ---------------------------------------------------------------------------
struct DncParams {
    const float *x4;
    const float *ctrl_Wih, *ctrl_bih, *ctrl_bhh;
    const float *w_key_W, *w_key_b, *w_beta_W, *w_beta_b;
    const float *w_alloc_W, *w_alloc_b, *w_gate_W, *w_gate_b;
    const float *w_erase_W, *w_erase_b, *w_add_W, *w_add_b;
    const float *r_key_W, *r_key_b, *r_beta_W, *r_beta_b;
    const float *r_mode_W, *r_mode_b;
    const float *out_W, *out_b, *lin_W, *lin_b;
    float *x5;
};

__device__ __forceinline__ float dot64(const float* a, const float* w) {
    float s = 0.0f;
#pragma unroll
    for (int k = 0; k < CTRL; ++k) s += a[k] * w[k];
    return s;
}

__global__ void __launch_bounds__(256) k_dnc(DncParams p)
{
    const int tid = threadIdx.x;
    __shared__ float x4s[HID], gs[4 * CTRL], hs[CTRL];
    __shared__ float mem[MEM_H * MEM_W], memnorm[MEM_H];
    __shared__ float wkey[MEM_W], wc_w[MEM_H], alloc[MEM_H], wl_w[MEM_H];
    __shared__ float ers[MEM_W], adds[MEM_W];
    __shared__ float rkey[N_READ * MEM_W], rbeta[N_READ], rkn[N_READ];
    __shared__ float modesraw[N_READ * 3], modes2[N_READ];
    __shared__ float wc_r[N_READ * MEM_H], rvec[N_READ * MEM_W];
    __shared__ float x4b[HID], scal[4];            // 0:wbeta 1:alloc_gate 2:write_gate 3:knorm

    if (tid < HID) x4s[tid] = p.x4[tid];
    mem[tid] = EPSF;                               // mem init 1e-6
    __syncthreads();

    // controller gates: g = x4 @ Wih[:, :20]^T + bih + bhh   (read_prev=0, h0=0)
    {
        float g = p.ctrl_bih[tid] + p.ctrl_bhh[tid];
#pragma unroll
        for (int k = 0; k < HID; ++k) g += x4s[k] * p.ctrl_Wih[tid * (HID + N_READ * MEM_W) + k];
        gs[tid] = g;
    }
    __syncthreads();
    if (tid < CTRL) {                              // c0=0: f-gate term drops
        float cc = fsig(gs[tid]) * ftanh(gs[2 * CTRL + tid]);
        float hh = fsig(gs[3 * CTRL + tid]) * ftanh(cc);
        hs[tid] = fminf(fmaxf(hh, -CLIPV), CLIPV);
    }
    __syncthreads();

    // head parameters (disjoint tid ranges, one phase)
    if (tid < MEM_W) {
        wkey[tid] = ftanh(dot64(hs, p.w_key_W  + tid * CTRL) + p.w_key_b[tid]);
        ers[tid]  = fsig (dot64(hs, p.w_erase_W + tid * CTRL) + p.w_erase_b[tid]);
        adds[tid] = ftanh(dot64(hs, p.w_add_W  + tid * CTRL) + p.w_add_b[tid]);
    } else if (tid >= 64 && tid < 64 + N_READ * MEM_W) {
        int r = tid - 64;
        rkey[r] = ftanh(dot64(hs, p.r_key_W + r * CTRL) + p.r_key_b[r]);
    } else if (tid >= 128 && tid < 128 + N_READ) {
        int r = tid - 128;
        rbeta[r] = fsoftp(dot64(hs, p.r_beta_W + r * CTRL) + p.r_beta_b[r]);
    } else if (tid >= 136 && tid < 136 + N_READ * 3) {
        int m = tid - 136;
        modesraw[m] = dot64(hs, p.r_mode_W + m * CTRL) + p.r_mode_b[m];
    } else if (tid == 160) scal[0] = fsoftp(dot64(hs, p.w_beta_W)  + p.w_beta_b[0]);
    else   if (tid == 161) scal[1] = fsig  (dot64(hs, p.w_alloc_W) + p.w_alloc_b[0]);
    else   if (tid == 162) scal[2] = fsig  (dot64(hs, p.w_gate_W)  + p.w_gate_b[0]);
    __syncthreads();

    // write-content weighting (pre-softmax sims in wc_w)
    if (tid < MEM_H) {
        float nn = 0.0f;
#pragma unroll
        for (int w = 0; w < MEM_W; ++w) nn += mem[tid * MEM_W + w] * mem[tid * MEM_W + w];
        memnorm[tid] = sqrtf(nn);
    } else if (tid == 16) {
        float nn = 0.0f;
#pragma unroll
        for (int w = 0; w < MEM_W; ++w) nn += wkey[w] * wkey[w];
        scal[3] = sqrtf(nn);
    }
    __syncthreads();
    if (tid < MEM_H) {
        float s = 0.0f;
#pragma unroll
        for (int w = 0; w < MEM_W; ++w)
            s += (wkey[w] / (scal[3] + EPSF)) * (mem[tid * MEM_W + w] / (memnorm[tid] + EPSF));
        wc_w[tid] = scal[0] * s;
    }
    __syncthreads();
    if (tid == 0) {                                // softmax + allocation (serial, 16 elems)
        float mx = wc_w[0];
        for (int n = 1; n < MEM_H; ++n) mx = fmaxf(mx, wc_w[n]);
        float sm = 0.0f;
        for (int n = 0; n < MEM_H; ++n) { wc_w[n] = __expf(wc_w[n] - mx); sm += wc_w[n]; }
        float rs = __builtin_amdgcn_rcpf(sm);
        for (int n = 0; n < MEM_H; ++n) wc_w[n] *= rs;
        // allocation on usage==0: u = EPS everywhere; stable ascending sort
        float u[MEM_H]; int ord[MEM_H];
        for (int n = 0; n < MEM_H; ++n) { u[n] = EPSF; ord[n] = n; }
        for (int i = 1; i < MEM_H; ++i) {          // stable insertion sort (strict <)
            int oi = ord[i]; float ui = u[oi]; int k = i - 1;
            while (k >= 0 && u[ord[k]] > ui) { ord[k + 1] = ord[k]; --k; }
            ord[k + 1] = oi;
        }
        float cp = 1.0f;
        for (int i = 0; i < MEM_H; ++i) {
            float su = u[ord[i]];
            alloc[ord[i]] = (1.0f - su) * cp;
            cp *= su;
        }
    }
    __syncthreads();
    if (tid < MEM_H)
        wl_w[tid] = scal[2] * (scal[1] * alloc[tid] + (1.0f - scal[1]) * wc_w[tid]);
    __syncthreads();

    // memory write (erase/add)
    {
        int n = tid >> 4, w = tid & 15;
        mem[tid] = mem[tid] * (1.0f - wl_w[n] * ers[w]) + wl_w[n] * adds[w];
    }
    __syncthreads();

    // read-content weighting on updated memory (fw=bw=0 since wl_r_prev=0)
    if (tid < MEM_H) {
        float nn = 0.0f;
#pragma unroll
        for (int w = 0; w < MEM_W; ++w) nn += mem[tid * MEM_W + w] * mem[tid * MEM_W + w];
        memnorm[tid] = sqrtf(nn);
    } else if (tid >= 16 && tid < 16 + N_READ) {
        int r = tid - 16; float nn = 0.0f;
#pragma unroll
        for (int w = 0; w < MEM_W; ++w) nn += rkey[r * MEM_W + w] * rkey[r * MEM_W + w];
        rkn[r] = sqrtf(nn);
    }
    __syncthreads();
    if (tid < N_READ * MEM_H) {
        int r = tid >> 4, n = tid & 15;
        float s = 0.0f;
#pragma unroll
        for (int w = 0; w < MEM_W; ++w)
            s += (rkey[r * MEM_W + w] / (rkn[r] + EPSF)) * (mem[n * MEM_W + w] / (memnorm[n] + EPSF));
        wc_r[tid] = rbeta[r] * s;
    }
    __syncthreads();
    if (tid < N_READ) {                            // softmax over 16 + mode softmax (content slot)
        int r = tid;
        float mx = wc_r[r * MEM_H];
        for (int n = 1; n < MEM_H; ++n) mx = fmaxf(mx, wc_r[r * MEM_H + n]);
        float sm = 0.0f;
        for (int n = 0; n < MEM_H; ++n) { float e = __expf(wc_r[r * MEM_H + n] - mx); wc_r[r * MEM_H + n] = e; sm += e; }
        float rs = __builtin_amdgcn_rcpf(sm);
        for (int n = 0; n < MEM_H; ++n) wc_r[r * MEM_H + n] *= rs;
        float m0 = modesraw[r * 3 + 0], m1 = modesraw[r * 3 + 1], m2 = modesraw[r * 3 + 2];
        float mm = fmaxf(m0, fmaxf(m1, m2));
        float e0 = __expf(m0 - mm), e1 = __expf(m1 - mm), e2 = __expf(m2 - mm);
        modes2[r] = e2 * __builtin_amdgcn_rcpf(e0 + e1 + e2);
    }
    __syncthreads();
    if (tid < N_READ * MEM_W) {                    // read vectors
        int r = tid >> 4, w = tid & 15;
        float s = 0.0f;
#pragma unroll
        for (int n = 0; n < MEM_H; ++n) s += modes2[r] * wc_r[r * MEM_H + n] * mem[n * MEM_W + w];
        rvec[r * MEM_W + w] = s;
    }
    __syncthreads();
    if (tid < HID) {                               // x4b = out([h, read_vec])
        float s = p.out_b[tid];
#pragma unroll
        for (int k = 0; k < CTRL; ++k) s += hs[k] * p.out_W[tid * (CTRL + N_READ * MEM_W) + k];
#pragma unroll
        for (int k = 0; k < N_READ * MEM_W; ++k) s += rvec[k] * p.out_W[tid * (CTRL + N_READ * MEM_W) + CTRL + k];
        x4b[tid] = s;
    }
    __syncthreads();
    if (tid < HID) {                               // x5 = relu(lin([x4, x4b]))
        float s = p.lin_b[tid];
#pragma unroll
        for (int k = 0; k < HID; ++k) s += x4s[k] * p.lin_W[tid * (2 * HID) + k];
#pragma unroll
        for (int k = 0; k < HID; ++k) s += x4b[k] * p.lin_W[tid * (2 * HID) + HID + k];
        p.x5[tid] = fmaxf(s, 0.0f);
    }
}

// ---------------------------------------------------------------------------
// K4: output GEMV  out[n] = act_W[n,:] . x5 + act_b[n]
// ---------------------------------------------------------------------------
__global__ void k_out(const float* __restrict__ x5, const float* __restrict__ actW,
                      const float* __restrict__ actb, float* __restrict__ out)
{
    __shared__ float xs[HID];
    if (threadIdx.x < HID) xs[threadIdx.x] = x5[threadIdx.x];
    __syncthreads();
    int n = blockIdx.x * blockDim.x + threadIdx.x;
    if (n < NUM_ACT) {
        float s = actb[n];
#pragma unroll
        for (int k = 0; k < HID; ++k) s += xs[k] * actW[n * HID + k];
        out[n] = s;
    }
}

// ---------------------------------------------------------------------------
extern "C" void kernel_launch(void* const* d_in, const int* in_sizes, int n_in,
                              void* d_out, int out_size, void* d_ws, size_t ws_size,
                              hipStream_t stream)
{
    const int*   x        = (const int*)  d_in[0];
    const float* emb      = (const float*)d_in[1];
    const float* lstm_Wih = (const float*)d_in[2];
    const float* lstm_Whh = (const float*)d_in[3];
    const float* lstm_bih = (const float*)d_in[4];
    const float* lstm_bhh = (const float*)d_in[5];
    const float* ctrl_Wih = (const float*)d_in[6];
    // d_in[7] ctrl_Whh unused (h0 == 0)
    const float* ctrl_bih = (const float*)d_in[8];
    const float* ctrl_bhh = (const float*)d_in[9];
    DncParams p;
    p.ctrl_Wih = ctrl_Wih; p.ctrl_bih = ctrl_bih; p.ctrl_bhh = ctrl_bhh;
    p.w_key_W   = (const float*)d_in[10]; p.w_key_b   = (const float*)d_in[11];
    p.w_beta_W  = (const float*)d_in[12]; p.w_beta_b  = (const float*)d_in[13];
    p.w_alloc_W = (const float*)d_in[14]; p.w_alloc_b = (const float*)d_in[15];
    p.w_gate_W  = (const float*)d_in[16]; p.w_gate_b  = (const float*)d_in[17];
    p.w_erase_W = (const float*)d_in[18]; p.w_erase_b = (const float*)d_in[19];
    p.w_add_W   = (const float*)d_in[20]; p.w_add_b   = (const float*)d_in[21];
    p.r_key_W   = (const float*)d_in[22]; p.r_key_b   = (const float*)d_in[23];
    p.r_beta_W  = (const float*)d_in[24]; p.r_beta_b  = (const float*)d_in[25];
    // d_in[26..27] r_free unused (usage == 0 regardless)
    p.r_mode_W  = (const float*)d_in[28]; p.r_mode_b  = (const float*)d_in[29];
    p.out_W     = (const float*)d_in[30]; p.out_b     = (const float*)d_in[31];
    p.lin_W     = (const float*)d_in[32]; p.lin_b     = (const float*)d_in[33];
    const float* act_W = (const float*)d_in[34];
    const float* act_b = (const float*)d_in[35];

    float* ws  = (float*)d_ws;
    float* pre = ws;                       // SEQ*GATES = 655360 floats (2.62 MB)
    float* x4  = ws + (size_t)SEQ * GATES; // 20 floats
    float* x5  = x4 + 32;                  // 20 floats
    p.x4 = x4; p.x5 = x5;

    // 512 time-tiles, 8 waves/block -> 64 blocks; 5 WMMAs per wave
    k_pregate<<<64, 256, 0, stream>>>(x, emb, lstm_Wih, lstm_bih, lstm_bhh, pre);
    k_lstm<<<1, 32, 0, stream>>>(pre, lstm_Whh, x4);
    k_dnc<<<1, 256, 0, stream>>>(p);
    k_out<<<(NUM_ACT + 255) / 256, 256, 0, stream>>>(x5, act_W, act_b, (float*)d_out);
}
